// PhysicsAwareModel_51144470561393
// MI455X (gfx1250) — compile-verified, matrix operand order checked by reading, not run on hardware
//
#include <hip/hip_runtime.h>
#include <math.h>

// Problem constants (match reference)
#define B_   32
#define S_   4096
#define DM_  256
#define DS_  16
#define LN_EPS 1e-5f

typedef unsigned int u32;
typedef __attribute__((ext_vector_type(16))) __bf16 v16bf;
typedef __attribute__((ext_vector_type(8)))  __bf16 v8bf;
typedef __attribute__((ext_vector_type(16))) float  v16f;
typedef __attribute__((ext_vector_type(8)))  float  v8f;
typedef __attribute__((ext_vector_type(2)))  float  v2f;
typedef __attribute__((ext_vector_type(4)))  u32    u32x4;
typedef __attribute__((ext_vector_type(8)))  int    i32x8;
typedef __attribute__((ext_vector_type(4)))  int    i32x4;

union V16BF { v16bf v; uint4 q[2]; u32 u[8]; };
union V8BF  { v8bf  v; uint4 q; };

// hardware packed f32->bf16 (RNE) via vector fptrunc -> v_cvt_pk_bf16_f32
__device__ __forceinline__ v16bf cvt16(const float4& f0, const float4& f1,
                                       const float4& f2, const float4& f3) {
    v16f xf;
    xf[0]=f0.x; xf[1]=f0.y; xf[2]=f0.z; xf[3]=f0.w;
    xf[4]=f1.x; xf[5]=f1.y; xf[6]=f1.z; xf[7]=f1.w;
    xf[8]=f2.x; xf[9]=f2.y; xf[10]=f2.z; xf[11]=f2.w;
    xf[12]=f3.x; xf[13]=f3.y; xf[14]=f3.z; xf[15]=f3.w;
    return __builtin_convertvector(xf, v16bf);
}

// ---------------------------------------------------------------------------
// K0: pack Wb (16x256) and Wg (256x256) into bf16 WMMA B-fragment order.
//   lane L (0..31): column n = L&15, k-half = L>>4
//   16 bf16 per lane: j<8  -> k = kt*32 + (L>>4)*8 + j
//                     j>=8 -> k = kt*32 + 16 + (L>>4)*8 + (j-8)
// grid: 17 blocks x 256. blocks 0..15 -> Wg n-tiles, block 16 -> Wb.
// ---------------------------------------------------------------------------
__global__ void __launch_bounds__(256)
pack_weights(const float* __restrict__ Wb, const float* __restrict__ Wg,
             u32* __restrict__ wbp, u32* __restrict__ wgp) {
    int t  = threadIdx.x;
    int kt = t >> 5, L = t & 31;
    int hf = L >> 4, nl = L & 15;
    int kb0 = kt * 32 + hf * 8;
    int kb1 = kt * 32 + 16 + hf * 8;
    const float* src;
    u32* dst;
    if ((int)blockIdx.x == 16) {
        src = Wb + nl * DM_;
        dst = wbp + (kt * 32 + L) * 8;
    } else {
        int nt = blockIdx.x;
        src = Wg + (nt * 16 + nl) * DM_;
        dst = wgp + ((nt * 8 + kt) * 32 + L) * 8;
    }
    float4 a0 = *(const float4*)(src + kb0);
    float4 a1 = *(const float4*)(src + kb0 + 4);
    float4 b0 = *(const float4*)(src + kb1);
    float4 b1 = *(const float4*)(src + kb1 + 4);
    V16BF c;
    c.v = cvt16(a0, a1, b0, b1);
    *(uint4*)(dst)     = c.q[0];
    *(uint4*)(dst + 4) = c.q[1];
}

// ---------------------------------------------------------------------------
// K1: Bx = x @ Wb^T + bb   (M=B*S=131072, N=16, K=256), bf16 WMMA, f32 accum.
// ---------------------------------------------------------------------------
__global__ void __launch_bounds__(256)
bx_gemm(const float* __restrict__ x, const float* __restrict__ bb,
        const u32* __restrict__ wbp, float* __restrict__ Bx) {
    int w = threadIdx.x >> 5, L = threadIdx.x & 31;
    int hf = L >> 4, nl = L & 15;
    int m0 = blockIdx.x * 128 + w * 16;
    int row = m0 + nl;  // A-fragment row for this lane
    v8f acc = {};
    #pragma unroll
    for (int kt = 0; kt < 8; ++kt) {
        int kb = kt * 32 + hf * 8;
        const float* xr = x + (size_t)row * DM_ + kb;
        float4 x0 = *(const float4*)(xr);
        float4 x1 = *(const float4*)(xr + 4);
        float4 x2 = *(const float4*)(xr + 16);
        float4 x3 = *(const float4*)(xr + 20);
        v16bf a = cvt16(x0, x1, x2, x3);
        V16BF b;
        const u32* bp = wbp + (kt * 32 + L) * 8;
        b.q[0] = *(const uint4*)(bp);
        b.q[1] = *(const uint4*)(bp + 4);
        acc = __builtin_amdgcn_wmma_f32_16x16x32_bf16(
            false, a, false, b.v, (short)0, acc, false, false);
    }
    float bias = bb[nl];
    #pragma unroll
    for (int j = 0; j < 8; ++j) {
        int r = j + 8 * hf;                 // C/D layout: M = j + 8*(lane>=16)
        Bx[(size_t)(m0 + r) * DS_ + nl] = acc[j] + bias;
    }
}

// ---------------------------------------------------------------------------
// K2: sequential scan, one wave per batch. Lane n owns state[n] (n<16).
// ---------------------------------------------------------------------------
__global__ void __launch_bounds__(32)
scan_kernel(const float* __restrict__ A, const float* __restrict__ Bx,
            float* __restrict__ states) {
    int b = blockIdx.x;
    int n = threadIdx.x;
    bool act = n < DS_;
    float a[16];
    #pragma unroll
    for (int k = 0; k < 16; ++k) a[k] = act ? A[n * DS_ + k] : 0.f;
    const float* bxp = Bx + (size_t)b * S_ * DS_;
    float* stp = states + (size_t)b * S_ * DS_;
    float s = 0.f;
    float bx = act ? bxp[n] : 0.f;
    for (int t = 0; t < S_; ++t) {
        float bx_next = (act && t + 1 < S_) ? bxp[(size_t)(t + 1) * DS_ + n] : 0.f;
        float acc = bx;
        #pragma unroll
        for (int k = 0; k < 16; ++k)
            acc = fmaf(a[k], __shfl(s, k, 32), acc);
        s = tanhf(acc);
        if (act) stp[(size_t)t * DS_ + n] = s;
        bx = bx_next;
    }
}

// ---------------------------------------------------------------------------
// K3: TDM-stage x tile -> LDS, then fused gate GEMM (bf16 WMMA) + y GEMM
//     (f32 WMMA, K=16) + silu gating + residual + LayerNorm.
//     One block = 16 rows (one batch, 16 timesteps), full DM=256 cols.
// ---------------------------------------------------------------------------
__global__ void __launch_bounds__(256)
fused_out(const float* __restrict__ x, const float* __restrict__ states,
          const u32* __restrict__ wgp,
          const float* __restrict__ Wc, const float* __restrict__ bc,
          const float* __restrict__ bg, const float* __restrict__ D,
          const float* __restrict__ gamma_, const float* __restrict__ beta_,
          float* __restrict__ out) {
    __shared__ float x_ls[16][DM_];      // x tile, f32 (TDM destination)
    __shared__ float h_ls[16][DM_];      // pre-LN result
    __shared__ float st_ls[16][DS_];     // states tile
    __shared__ u32  a_frag[8][32][8];    // x tile as bf16 A-fragments

    int tid = threadIdx.x;
    int m0 = blockIdx.x * 16;            // flat row base (b*S + t0)
    int w = tid >> 5, L = tid & 31, hf = L >> 4, nl = L & 15;

    // ---- stage states (all threads) ----
    st_ls[tid >> 4][tid & 15] =
        states[(size_t)(m0 + (tid >> 4)) * DS_ + (tid & 15)];

    // ---- TDM: DMA the 16x256 f32 x tile into LDS (wave 0 issues) ----
    if (w == 0) {
        unsigned long long ga =
            (unsigned long long)(const void*)(x + (size_t)m0 * DM_);
        u32 lds_addr = (u32)(size_t)&x_ls[0][0];
        u32x4 g0;
        g0[0] = 1u;                                  // count=1, user descriptor
        g0[1] = lds_addr;                            // bits 63:32  lds_addr
        g0[2] = (u32)ga;                             // bits 95:64  global_addr lo
        g0[3] = (u32)(ga >> 32) | (2u << 30);        // ga hi | type=2 ("image")
        i32x8 g1;
        g1[0] = 2 << 16;        // workgroup_mask=0, data_size=2 (4 bytes)
        g1[1] = DM_ << 16;      // tensor_dim0[15:0]=256
        g1[2] = 16 << 16;       // tensor_dim1[15:0]=16
        g1[3] = DM_ << 16;      // tensor_dim1 hi=0 | tile_dim0=256
        g1[4] = 16;             // tile_dim1=16, tile_dim2=0
        g1[5] = DM_;            // tensor_dim0_stride=256 (low 32 of 48)
        g1[6] = 0;              // stride hi, tensor_dim1_stride lo
        g1[7] = 0;
        i32x4 z4 = {};
        i32x8 z8 = {};
        __builtin_amdgcn_tensor_load_to_lds(g0, g1, z4, z4, z8, 0);
        __builtin_amdgcn_s_wait_tensorcnt(0);
    }
    __syncthreads();

    // ---- convert x tile (LDS f32) into bf16 A-fragments ----
    {
        int r = tid >> 4, c0 = (tid & 15) * 16;
        float4 f0 = *(const float4*)&x_ls[r][c0];
        float4 f1 = *(const float4*)&x_ls[r][c0 + 4];
        float4 f2 = *(const float4*)&x_ls[r][c0 + 8];
        float4 f3 = *(const float4*)&x_ls[r][c0 + 12];
        // kr=k&31: kr<8 -> lane r; 8..15 -> lane r+16; 16..31 -> dwords 4..7
        int kt = c0 >> 5;
        int dwoff = (c0 & 16) ? 4 : 0;
        V16BF cv;
        cv.v = cvt16(f0, f1, f2, f3);
        *(uint4*)&a_frag[kt][r][dwoff]      = cv.q[0];
        *(uint4*)&a_frag[kt][r + 16][dwoff] = cv.q[1];
    }
    __syncthreads();

    #pragma unroll
    for (int it = 0; it < 2; ++it) {
        int nt = w * 2 + it;
        int n0 = nt * 16;
        // ---- gate GEMM: (16 x 256) x (256 x 16), bf16, 8 chained WMMAs ----
        v8f gacc = {};
        #pragma unroll
        for (int kt = 0; kt < 8; ++kt) {
            V16BF a, b;
            a.q[0] = *(const uint4*)&a_frag[kt][L][0];
            a.q[1] = *(const uint4*)&a_frag[kt][L][4];
            const u32* bp = wgp + ((nt * 8 + kt) * 32 + L) * 8;
            b.q[0] = *(const uint4*)(bp);
            b.q[1] = *(const uint4*)(bp + 4);
            gacc = __builtin_amdgcn_wmma_f32_16x16x32_bf16(
                false, a.v, false, b.v, (short)0, gacc, false, false);
        }
        // ---- y GEMM: states (16x16) x Wc^T (16x16), exact f32, 4x 16x16x4 ----
        v8f yacc = {};
        #pragma unroll
        for (int kt = 0; kt < 4; ++kt) {
            int k0 = kt * 4 + hf * 2;    // f32 A/B frag: lanes>=16 carry K+2,K+3
            v2f av, bv;
            av[0] = st_ls[nl][k0];
            av[1] = st_ls[nl][k0 + 1];
            const float* wcr = Wc + (n0 + nl) * DS_ + k0;
            bv[0] = wcr[0];
            bv[1] = wcr[1];
            yacc = __builtin_amdgcn_wmma_f32_16x16x4_f32(
                false, av, false, bv, (short)0, yacc, false, false);
        }
        // ---- gate/y tiles element-aligned: fuse silu+gating+residual ----
        int c = n0 + nl;
        float bgv = bg[c], bcv = bc[c], Dv = D[c];
        #pragma unroll
        for (int j = 0; j < 8; ++j) {
            int r = j + 8 * hf;
            float xv = x_ls[r][c];
            float g0v = gacc[j] + bgv;
            float g  = g0v / (1.f + __expf(-g0v));     // silu
            float y  = yacc[j] + bcv + Dv * xv;
            float o  = g * y + (1.f - g) * xv;
            h_ls[r][c] = o + xv;                        // h = out + x
        }
    }
    __syncthreads();

    // ---- LayerNorm over DM per row: 16 lanes/row, half-wave shfl reduce ----
    {
        int r = tid >> 4, c0 = (tid & 15) * 16;
        float hbuf[16];
        float sum = 0.f, sq = 0.f;
        #pragma unroll
        for (int i = 0; i < 16; ++i) {
            float hv = h_ls[r][c0 + i];
            hbuf[i] = hv;
            sum += hv;
            sq  += hv * hv;
        }
        #pragma unroll
        for (int m = 1; m < 16; m <<= 1) {
            sum += __shfl_xor(sum, m, 32);
            sq  += __shfl_xor(sq, m, 32);
        }
        float mu  = sum * (1.f / DM_);
        float var = sq * (1.f / DM_) - mu * mu;
        float rs  = rsqrtf(var + LN_EPS);
        float* op = out + (size_t)(m0 + r) * DM_ + c0;
        #pragma unroll
        for (int i = 0; i < 16; ++i)
            op[i] = (hbuf[i] - mu) * rs * gamma_[c0 + i] + beta_[c0 + i];
    }
}

// ---------------------------------------------------------------------------
// Host-side launcher. Inputs: x, A, Wb, bb, Wc, bc, Wg, bg, D, gamma, beta.
// Workspace layout: [wgp bf16-packed 128KB][wbp 8KB][Bx 8MB][states 8MB]
// ---------------------------------------------------------------------------
extern "C" void kernel_launch(void* const* d_in, const int* in_sizes, int n_in,
                              void* d_out, int out_size, void* d_ws, size_t ws_size,
                              hipStream_t stream) {
    const float* x  = (const float*)d_in[0];
    const float* A  = (const float*)d_in[1];
    const float* Wb = (const float*)d_in[2];
    const float* bb = (const float*)d_in[3];
    const float* Wc = (const float*)d_in[4];
    const float* bc = (const float*)d_in[5];
    const float* Wg = (const float*)d_in[6];
    const float* bg = (const float*)d_in[7];
    const float* D  = (const float*)d_in[8];
    const float* gm = (const float*)d_in[9];
    const float* bt = (const float*)d_in[10];
    float* out = (float*)d_out;

    char* ws = (char*)d_ws;
    u32*   wgp    = (u32*)(ws);                       // 16*8*32*8*4 = 131072 B
    u32*   wbp    = (u32*)(ws + 131072);              // 8*32*8*4    = 8192 B
    float* Bx     = (float*)(ws + 139264);            // 131072*16*4 = 8 MB
    float* states = (float*)(ws + 139264 + 8388608);  // 8 MB

    pack_weights<<<17,   256, 0, stream>>>(Wb, Wg, wbp, wgp);
    bx_gemm     <<<1024, 256, 0, stream>>>(x, bb, wbp, Bx);
    scan_kernel <<<B_,    32, 0, stream>>>(A, Bx, states);
    fused_out   <<<(B_ * S_) / 16, 256, 0, stream>>>(x, states, wgp, Wc, bc,
                                                     bg, D, gm, bt, out);
}
